// ForgettingAttentionLayer_64338610094608
// MI455X (gfx1250) — matrix operand order repaired and checked by reading, not compiled, not run on hardware
//
#include <hip/hip_runtime.h>
#include <cstdint>
#include <cstddef>

#define B_ 2
#define T_ 2048
#define C_ 2048
#define H_ 16
#define D_ 128
#define M_ (B_*T_)   /* 4096 rows */

typedef __attribute__((ext_vector_type(16))) __bf16 v16bf;
typedef __attribute__((ext_vector_type(8)))  __bf16 v8bf;
typedef __attribute__((ext_vector_type(8)))  float  v8f;
typedef __attribute__((ext_vector_type(4)))  unsigned int v4u;
typedef __attribute__((ext_vector_type(8)))  int    v8i_;
typedef __attribute__((ext_vector_type(4)))  int    v4i_;

static __device__ __forceinline__ v8f vzero8() {
  v8f z;
#pragma unroll
  for (int i = 0; i < 8; ++i) z[i] = 0.f;
  return z;
}

// A-matrix 16x32 bf16 fragment, contiguous k order (matches B loads):
// lanes 0-15 hold K=0..15; lanes 16-31 hold K=16..31.
static __device__ __forceinline__ v16bf load_a_frag(const __bf16* rowptr, int half) {
  v8bf lo = *(const v8bf*)(rowptr + 16 * half);
  v8bf hi = *(const v8bf*)(rowptr + 16 * half + 8);
  v16bf r;
#pragma unroll
  for (int i = 0; i < 8; ++i) { r[i] = lo[i]; r[8 + i] = hi[i]; }
  return r;
}

// ---------------------------------------------------------------------------
// TDM: 2D bf16 tile DMA global -> LDS.  Builds D# group0/group1 per CDNA5 ISA
// (8.3/8.4): count=1, type=2 (image), data_size=2B, tile rows compacted into
// LDS at lds_addr (row-major [tile_d1][tile_d0]).  This toolchain uses the
// 6-arg builtin: (g0 u32x4, g1 i32x8, g2 i32x4, g3 i32x4, g4 i32x8, cpol).
// ---------------------------------------------------------------------------
static __device__ __forceinline__ void tdm_load_2d_bf16(
    unsigned lds_addr, const void* gaddr, unsigned tensor_d0, unsigned tensor_d1,
    unsigned tile_d0, unsigned tile_d1, unsigned stride0) {
  const unsigned long long ga = (unsigned long long)(uintptr_t)gaddr;
  v4u g0;
  g0[0] = 1u;                                   // count=1 (user mode)
  g0[1] = lds_addr;                             // lds_addr [63:32]
  g0[2] = (unsigned)ga;                         // global_addr [95:64]
  g0[3] = (unsigned)(ga >> 32) | 0x80000000u;   // global_addr[56:32] | type=2
  v8i_ g1;
  g1[0] = (int)(1u << 16);                                    // data_size=2B
  g1[1] = (int)((tensor_d0 & 0xFFFFu) << 16);                 // tensor_dim0 lo
  g1[2] = (int)((tensor_d0 >> 16) | ((tensor_d1 & 0xFFFFu) << 16));
  g1[3] = (int)((tensor_d1 >> 16) | (tile_d0 << 16));         // tile_dim0
  g1[4] = (int)(tile_d1 & 0xFFFFu);                           // tile_dim1
  g1[5] = (int)stride0;                                       // dim0 stride lo
  g1[6] = 0;
  g1[7] = 0;
  v4i_ z4 = {0, 0, 0, 0};
  v8i_ z8;
#pragma unroll
  for (int i = 0; i < 8; ++i) z8[i] = 0;
  __builtin_amdgcn_tensor_load_to_lds(g0, g1, z4, z4, z8, 0);
}

// ---------------------------------------------------------------------------
// f32 -> bf16 pack
// ---------------------------------------------------------------------------
__global__ __launch_bounds__(256) void pack_bf16(const float* __restrict__ in,
                                                 __bf16* __restrict__ out, int n) {
  for (int i = blockIdx.x * 256 + threadIdx.x; i < n; i += gridDim.x * 256)
    out[i] = (__bf16)in[i];
}

// ---------------------------------------------------------------------------
// C[M,N] = A[M,K](bf16,row-major) * W[N,K](bf16,row-major)^T   (f32 out)
// Block: 256 threads (8 waves), tile 128x64, K-step 32; tiles staged into LDS
// by the Tensor Data Mover (wave 0), waves read WMMA fragments from LDS.
// ---------------------------------------------------------------------------
__global__ __launch_bounds__(256) void gemm_bf16_nt(const __bf16* __restrict__ A,
                                                    const __bf16* __restrict__ W,
                                                    float* __restrict__ Cout,
                                                    int M, int N, int K) {
  __shared__ __bf16 As[128 * 32];   // 8 KB
  __shared__ __bf16 Bs[64 * 32];    // 4 KB
  const int tid  = threadIdx.x;
  const int wave = tid >> 5, lane = tid & 31;
  const int half = lane >> 4, l16 = lane & 15;
  const int m0 = blockIdx.y * 128;
  const int n0 = blockIdx.x * 64;
  const int wm = (wave & 3) * 32;
  const int wn = (wave >> 2) * 32;
  const unsigned ldsA = (unsigned)(uintptr_t)&As[0];
  const unsigned ldsB = (unsigned)(uintptr_t)&Bs[0];

  v8f acc[2][2];
#pragma unroll
  for (int a = 0; a < 2; ++a)
#pragma unroll
    for (int b = 0; b < 2; ++b) acc[a][b] = vzero8();

  const int nkt = K >> 5;
  for (int kt = 0; kt < nkt; ++kt) {
    const int k0 = kt << 5;
    if (wave == 0) {
      // DMA the A (128x32) and B (64x32) tiles into LDS; wait on TENSORcnt
      // before the block barrier so all waves see the staged data.
      tdm_load_2d_bf16(ldsA, &A[(size_t)m0 * K + k0], (unsigned)K, (unsigned)M,
                       32u, 128u, (unsigned)K);
      tdm_load_2d_bf16(ldsB, &W[(size_t)n0 * K + k0], (unsigned)K, (unsigned)N,
                       32u, 64u, (unsigned)K);
      __builtin_amdgcn_s_wait_tensorcnt(0);
    } else if (wave == 1 && kt + 1 < nkt) {
      __builtin_prefetch(&A[(size_t)(m0 + lane * 4) * K + k0 + 32], 0, 1);
      __builtin_prefetch(&W[(size_t)(n0 + lane * 2) * K + k0 + 32], 0, 1);
    }
    __syncthreads();

    v16bf af[2], bw[2];
#pragma unroll
    for (int mi = 0; mi < 2; ++mi)
      af[mi] = load_a_frag(&As[(wm + mi * 16 + l16) * 32], half);
#pragma unroll
    for (int ni = 0; ni < 2; ++ni)
      bw[ni] = *(const v16bf*)&Bs[(wn + ni * 16 + l16) * 32 + half * 16];
#pragma unroll
    for (int mi = 0; mi < 2; ++mi)
#pragma unroll
      for (int ni = 0; ni < 2; ++ni)
        acc[mi][ni] = __builtin_amdgcn_wmma_f32_16x16x32_bf16(
            false, af[mi], false, bw[ni], (short)0, acc[mi][ni], false, false);
    __syncthreads();
  }

#pragma unroll
  for (int mi = 0; mi < 2; ++mi)
#pragma unroll
    for (int ni = 0; ni < 2; ++ni)
#pragma unroll
      for (int i = 0; i < 8; ++i) {
        const int r = m0 + wm + mi * 16 + i + 8 * half;
        const int c = n0 + wn + ni * 16 + l16;
        Cout[(size_t)r * N + c] = acc[mi][ni][i];
      }
}

// ---------------------------------------------------------------------------
// Gates: pq/pk/pv = sigmoid(x @ Ws^T); logf = log_sigmoid(x @ Wf^T + bf)
// One block per (b,t) row; x row staged into LDS with async global->LDS loads.
// ---------------------------------------------------------------------------
__global__ __launch_bounds__(256) void gates_kernel(
    const float* __restrict__ x, const float* __restrict__ Wsq,
    const float* __restrict__ Wsk, const float* __restrict__ Wsv,
    const float* __restrict__ Wf, const float* __restrict__ bf,
    float* __restrict__ pq, float* __restrict__ pk, float* __restrict__ pv,
    float* __restrict__ lf) {
  __shared__ float xs[C_];
  const int row = blockIdx.x;  // b*T + t
  const int tid = threadIdx.x;
  {
    // each thread DMAs 32 bytes (8 floats) of the x row into LDS
    const int i = tid * 8;
    const unsigned l0 = (unsigned)(uintptr_t)&xs[i];
    const unsigned long long g0 =
        (unsigned long long)(uintptr_t)&x[(size_t)row * C_ + i];
    asm volatile("global_load_async_to_lds_b128 %0, %1, off"
                 :: "v"(l0), "v"(g0) : "memory");
    asm volatile("global_load_async_to_lds_b128 %0, %1, off offset:16"
                 :: "v"(l0), "v"(g0) : "memory");
    asm volatile("s_wait_asynccnt 0x0" ::: "memory");
  }
  __syncthreads();
  const int oid = tid >> 2, part = tid & 3;   // 64 outputs x 4 partials
  const int set = oid >> 4, h = oid & 15;
  const float* Wrow =
      (set == 0 ? Wsq : set == 1 ? Wsk : set == 2 ? Wsv : Wf) + (size_t)h * C_;
  float acc = 0.f;
  for (int k = part * 512; k < part * 512 + 512; ++k) acc += xs[k] * Wrow[k];
  acc += __shfl_xor(acc, 1, 32);
  acc += __shfl_xor(acc, 2, 32);
  if (part == 0) {
    const size_t o = (size_t)row * H_ + h;
    if (set == 3) {
      const float z = acc + bf[h];
      lf[o] = fminf(z, 0.f) - log1pf(__expf(-fabsf(z)));
    } else {
      const float s = 1.f / (1.f + __expf(-acc));
      (set == 0 ? pq : set == 1 ? pk : pv)[o] = s;
    }
  }
}

// ---------------------------------------------------------------------------
// cum[b,h,t] = inclusive prefix sum over t of lf[b,t,h].  One wave per (b,h).
// ---------------------------------------------------------------------------
__global__ __launch_bounds__(32) void cumsum_kernel(const float* __restrict__ lf,
                                                    float* __restrict__ cum) {
  const int bh = blockIdx.x;           // b*H + h
  const int b = bh / H_, h = bh % H_;
  const int lane = threadIdx.x;
  float carry = 0.f;
  for (int t0 = 0; t0 < T_; t0 += 32) {
    float v = lf[((size_t)b * T_ + t0 + lane) * H_ + h];
#pragma unroll
    for (int off = 1; off < 32; off <<= 1) {
      float n = __shfl_up(v, off, 32);
      if (lane >= off) v += n;
    }
    cum[(size_t)bh * T_ + t0 + lane] = v + carry;
    carry += __shfl(v, 31, 32);
  }
}

// ---------------------------------------------------------------------------
// Shift-mix + RoPE.  Reads raw projections (b*T+t, h*D+d) f32, gates (b,t,h);
// writes bf16 q/k/v in (b,h,t,d) layout.  One block per (b,t,h), thread = d.
// ---------------------------------------------------------------------------
__global__ __launch_bounds__(128) void shift_rope_kernel(
    const float* __restrict__ qr, const float* __restrict__ kr,
    const float* __restrict__ vr, const float* __restrict__ pq,
    const float* __restrict__ pk, const float* __restrict__ pv,
    __bf16* __restrict__ qo, __bf16* __restrict__ ko, __bf16* __restrict__ vo) {
  const int h = blockIdx.x, t = blockIdx.y, b = blockIdx.z;
  const int d = threadIdx.x;
  const int dp = d ^ 64;               // rotate partner
  const size_t row = (size_t)(b * T_ + t);
  const size_t i0 = row * C_ + h * D_;
  const size_t g = row * H_ + h;
  const float gq = pq[g], gk = pk[g], gv = pv[g];

  float yq = qr[i0 + d], yqp = qr[i0 + dp];
  float yk = kr[i0 + d], ykp = kr[i0 + dp];
  float yv = vr[i0 + d];
  float zq = 0.f, zqp = 0.f, zk = 0.f, zkp = 0.f, zv = 0.f;
  if (t > 0) {
    const size_t ip = i0 - C_;
    zq = qr[ip + d]; zqp = qr[ip + dp];
    zk = kr[ip + d]; zkp = kr[ip + dp];
    zv = vr[ip + d];
  }
  const float mq  = yq  * (1.f - gq) + zq  * gq;
  const float mqp = yqp * (1.f - gq) + zqp * gq;
  const float mk  = yk  * (1.f - gk) + zk  * gk;
  const float mkp = ykp * (1.f - gk) + zkp * gk;
  const float mv  = yv  * (1.f - gv) + zv  * gv;

  const int j = d & 63;
  float sn, cs;
  // inv_freq = 10000^(-2j/128) = exp(-j * ln(10000)*2/128)
  __sincosf((float)t * __expf(-0.14391157f * (float)j), &sn, &cs);
  const float sgn = (d < 64) ? 1.f : -1.f;
  const float qout = mq * cs + sgn * mqp * sn;
  const float kout = mk * cs + sgn * mkp * sn;

  const size_t o = ((size_t)(b * H_ + h) * T_ + t) * D_ + d;
  qo[o] = (__bf16)qout;
  ko[o] = (__bf16)kout;
  vo[o] = (__bf16)mv;
}

// ---------------------------------------------------------------------------
// Flash-style causal attention with forgetting decay.
// Block: 128 threads (4 waves), 64 query rows; wave w owns rows w*16..w*16+15.
// Key tiles of 32; S = Q K^T via WMMA, online softmax, P V via WMMA with
// V transposed into LDS. Decay bias: + cum[t] - cum[s].
// ---------------------------------------------------------------------------
__global__ __launch_bounds__(128) void attn_kernel(
    const __bf16* __restrict__ Q, const __bf16* __restrict__ Kk,
    const __bf16* __restrict__ V, const float* __restrict__ cum,
    __bf16* __restrict__ O) {
  const int bh = blockIdx.y;            // b*H + h
  const int qb = blockIdx.x * 64;
  const int tid = threadIdx.x;
  const int wave = tid >> 5, lane = tid & 31;
  const int half = lane >> 4, l16 = lane & 15;

  __shared__ __bf16 vt[128 * 32];       // V^T tile: [d][s], 8 KB
  __shared__ __bf16 pls[4][16 * 32];    // per-wave P strip, 4 KB

  const size_t base = (size_t)bh * T_ * D_;
  const int tq0 = qb + wave * 16;

  // Preload Q fragments for this wave's 16 rows (full D=128 -> 4 K-steps).
  v16bf aq[4];
#pragma unroll
  for (int kd = 0; kd < 4; ++kd)
    aq[kd] = load_a_frag(&Q[base + (size_t)(tq0 + l16) * D_ + kd * 32], half);

  float cum_t[8];
  int trow[8];
#pragma unroll
  for (int i = 0; i < 8; ++i) {
    trow[i] = tq0 + i + 8 * half;
    cum_t[i] = cum[(size_t)bh * T_ + trow[i]];
  }

  float m_i[8], l_i[8];
  v8f o_acc[8];
#pragma unroll
  for (int i = 0; i < 8; ++i) { m_i[i] = -1e30f; l_i[i] = 0.f; }
#pragma unroll
  for (int nt = 0; nt < 8; ++nt) o_acc[nt] = vzero8();

  const int nkt = (qb + 64) >> 5;
  for (int kt = 0; kt < nkt; ++kt) {
    const int s0 = kt << 5;
    // Stage V^T: 32 keys x 128 d  ->  vt[d][key]
    {
      const int key = tid & 31, db = (tid >> 5) * 32;
      const __bf16* vrow = &V[base + (size_t)(s0 + key) * D_];
#pragma unroll
      for (int i = 0; i < 32; ++i) vt[(db + i) * 32 + key] = vrow[db + i];
    }
    __syncthreads();

    // S strip: 16 rows x 32 keys (2 accumulator tiles)
    v8f s_acc[2];
#pragma unroll
    for (int nt = 0; nt < 2; ++nt) {
      s_acc[nt] = vzero8();
#pragma unroll
      for (int kd = 0; kd < 4; ++kd) {
        v16bf bk = *(const v16bf*)&Kk[base + (size_t)(s0 + nt * 16 + l16) * D_ +
                                      kd * 32 + half * 16];
        s_acc[nt] = __builtin_amdgcn_wmma_f32_16x16x32_bf16(
            false, aq[kd], false, bk, (short)0, s_acc[nt], false, false);
      }
    }

    // scale + decay + causal mask
    float sv[2][8];
#pragma unroll
    for (int nt = 0; nt < 2; ++nt) {
      const int scol = s0 + nt * 16 + l16;
      const float cs = cum[(size_t)bh * T_ + scol];
#pragma unroll
      for (int i = 0; i < 8; ++i) {
        float v = s_acc[nt][i] * 0.08838834764831845f + cum_t[i] - cs;
        if (scol > trow[i]) v = -1e30f;
        sv[nt][i] = v;
      }
    }

    // online softmax: row max / exp / row sum (rows live per lane-half)
    float mnew[8], rsum[8];
#pragma unroll
    for (int i = 0; i < 8; ++i) {
      float v = fmaxf(sv[0][i], sv[1][i]);
#pragma unroll
      for (int m = 1; m < 16; m <<= 1) v = fmaxf(v, __shfl_xor(v, m, 32));
      mnew[i] = fmaxf(m_i[i], v);
    }
#pragma unroll
    for (int i = 0; i < 8; ++i) {
      const float p0 = __expf(sv[0][i] - mnew[i]);
      const float p1 = __expf(sv[1][i] - mnew[i]);
      const int prow = i + 8 * half;
      pls[wave][prow * 32 + l16]      = (__bf16)p0;
      pls[wave][prow * 32 + 16 + l16] = (__bf16)p1;
      float v = p0 + p1;
#pragma unroll
      for (int m = 1; m < 16; m <<= 1) v += __shfl_xor(v, m, 32);
      rsum[i] = v;
    }
#pragma unroll
    for (int i = 0; i < 8; ++i) {
      const float alpha = __expf(m_i[i] - mnew[i]);
      l_i[i] = l_i[i] * alpha + rsum[i];
      m_i[i] = mnew[i];
#pragma unroll
      for (int nt = 0; nt < 8; ++nt) o_acc[nt][i] *= alpha;
    }

    // O += P (16x32) * V (32x128)   (LDS ops from same wave are in-order)
    const v16bf ap = load_a_frag(&pls[wave][l16 * 32], half);
#pragma unroll
    for (int nt = 0; nt < 8; ++nt) {
      const v16bf bv = *(const v16bf*)&vt[(nt * 16 + l16) * 32 + half * 16];
      o_acc[nt] = __builtin_amdgcn_wmma_f32_16x16x32_bf16(
          false, ap, false, bv, (short)0, o_acc[nt], false, false);
    }
    __syncthreads();
  }

  // Normalize and store O in (b*T+t, h*D+d) bf16 layout for the output GEMM.
  const int b = bh / H_, h = bh % H_;
#pragma unroll
  for (int nt = 0; nt < 8; ++nt)
#pragma unroll
    for (int i = 0; i < 8; ++i) {
      const int t = trow[i];
      const int d = nt * 16 + l16;
      O[((size_t)(b * T_ + t)) * C_ + h * D_ + d] = (__bf16)(o_acc[nt][i] / l_i[i]);
    }
}

// ---------------------------------------------------------------------------
// Host launch
// ---------------------------------------------------------------------------
extern "C" void kernel_launch(void* const* d_in, const int* in_sizes, int n_in,
                              void* d_out, int out_size, void* d_ws, size_t ws_size,
                              hipStream_t stream) {
  const float* x   = (const float*)d_in[0];
  const float* Wq  = (const float*)d_in[1];
  const float* Wsq = (const float*)d_in[2];
  const float* Wk  = (const float*)d_in[3];
  const float* Wsk = (const float*)d_in[4];
  const float* Wv  = (const float*)d_in[5];
  const float* Wsv = (const float*)d_in[6];
  const float* Wf  = (const float*)d_in[7];
  const float* bf  = (const float*)d_in[8];
  const float* Wo  = (const float*)d_in[9];

  char* ws = (char*)d_ws;
  size_t off = 0;
  auto carve = [&](size_t bytes) -> char* {
    char* p = ws + off;
    off = (off + bytes + 255) & ~(size_t)255;
    return p;
  };
  __bf16* xbf  = (__bf16*)carve((size_t)M_ * C_ * 2);
  __bf16* wqbf = (__bf16*)carve((size_t)C_ * C_ * 2);
  __bf16* wkbf = (__bf16*)carve((size_t)C_ * C_ * 2);
  __bf16* wvbf = (__bf16*)carve((size_t)C_ * C_ * 2);
  __bf16* wobf = (__bf16*)carve((size_t)C_ * C_ * 2);
  float*  qraw = (float*)carve((size_t)M_ * C_ * 4);
  float*  kraw = (float*)carve((size_t)M_ * C_ * 4);
  float*  vraw = (float*)carve((size_t)M_ * C_ * 4);
  float*  pq   = (float*)carve((size_t)M_ * H_ * 4);
  float*  pk   = (float*)carve((size_t)M_ * H_ * 4);
  float*  pv   = (float*)carve((size_t)M_ * H_ * 4);
  float*  lf   = (float*)carve((size_t)M_ * H_ * 4);
  float*  cum  = (float*)carve((size_t)B_ * H_ * T_ * 4);
  __bf16* qbf  = (__bf16*)carve((size_t)B_ * H_ * T_ * D_ * 2);
  __bf16* kbf  = (__bf16*)carve((size_t)B_ * H_ * T_ * D_ * 2);
  __bf16* vbf  = (__bf16*)carve((size_t)B_ * H_ * T_ * D_ * 2);
  __bf16* obf  = (__bf16*)carve((size_t)M_ * C_ * 2);
  (void)ws_size; (void)in_sizes; (void)n_in; (void)out_size;

  // 1) pack to bf16
  pack_bf16<<<4096, 256, 0, stream>>>(x,  xbf,  M_ * C_);
  pack_bf16<<<2048, 256, 0, stream>>>(Wq, wqbf, C_ * C_);
  pack_bf16<<<2048, 256, 0, stream>>>(Wk, wkbf, C_ * C_);
  pack_bf16<<<2048, 256, 0, stream>>>(Wv, wvbf, C_ * C_);
  pack_bf16<<<2048, 256, 0, stream>>>(Wo, wobf, C_ * C_);

  // 2) q/k/v projections (WMMA GEMM, TDM-staged tiles)
  const dim3 gg(C_ / 64, M_ / 128);
  gemm_bf16_nt<<<gg, 256, 0, stream>>>(xbf, wqbf, qraw, M_, C_, C_);
  gemm_bf16_nt<<<gg, 256, 0, stream>>>(xbf, wkbf, kraw, M_, C_, C_);
  gemm_bf16_nt<<<gg, 256, 0, stream>>>(xbf, wvbf, vraw, M_, C_, C_);

  // 3) gates + log-forget (async global->LDS staging)
  gates_kernel<<<M_, 256, 0, stream>>>(x, Wsq, Wsk, Wsv, Wf, bf, pq, pk, pv, lf);

  // 4) forget cumsum
  cumsum_kernel<<<B_ * H_, 32, 0, stream>>>(lf, cum);

  // 5) shift-mix + RoPE -> bf16 (b,h,t,d)
  shift_rope_kernel<<<dim3(H_, T_, B_), 128, 0, stream>>>(
      qraw, kraw, vraw, pq, pk, pv, qbf, kbf, vbf);

  // 6) causal decay attention
  attn_kernel<<<dim3(T_ / 64, B_ * H_), 128, 0, stream>>>(qbf, kbf, vbf, cum, obf);

  // 7) output projection -> d_out (f32)
  gemm_bf16_nt<<<gg, 256, 0, stream>>>(obf, wobf, (float*)d_out, M_, C_, C_);
}
